// Attention_3908420239425
// MI455X (gfx1250) — matrix operand order repaired and checked by reading, not compiled, not run on hardware
//
#include <hip/hip_runtime.h>
#include <hip/hip_bf16.h>

typedef __attribute__((ext_vector_type(16))) _Float16 v16h;
typedef __attribute__((ext_vector_type(8)))  _Float16 v8h;
typedef __attribute__((ext_vector_type(8)))  float    v8f;

#define WMMA_F16(a, b, c) \
  __builtin_amdgcn_wmma_f32_16x16x32_f16(false, (a), false, (b), (short)0, (c), false, false)

// ---------------------------------------------------------------------------
// Fragment loaders (wave32).
// A fragment 16x32 f16, row-major source, leading dim ld:
//   lane L: row = L&15, elements 0..7 -> K = 8*(L>>4)+0..7, elements 8..15 -> +16
static __device__ __forceinline__ v16h load_frag_A_g(const _Float16* __restrict__ base, int ld) {
  const int L = threadIdx.x & 31;
  const _Float16* p = base + (L & 15) * ld + ((L >> 4) << 3);
  v8h lo = *(const v8h*)(p);
  v8h hi = *(const v8h*)(p + 16);
  return __builtin_shufflevector(lo, hi, 0,1,2,3,4,5,6,7,8,9,10,11,12,13,14,15);
}

// B fragment 32x16 f16 where element [k][n] = base[n*ld + k] (the N index walks
// rows of memory -> contiguous K loads). lane L: n = L&15,
// elements 0..7 -> K = 8*(L>>4) + 0..7, elements 8..15 -> +16
static __device__ __forceinline__ v16h load_frag_B_g(const _Float16* __restrict__ base, int ld) {
  const int L = threadIdx.x & 31;
  const _Float16* p = base + (L & 15) * ld + ((L >> 4) << 3);
  v8h lo = *(const v8h*)(p);
  v8h hi = *(const v8h*)(p + 16);
  return __builtin_shufflevector(lo, hi, 0,1,2,3,4,5,6,7,8,9,10,11,12,13,14,15);
}

// ---------------------------------------------------------------------------
// Conversion / transpose helpers (one-time, everything stays L2-resident)
__global__ __launch_bounds__(256) void k_cvt_f32_f16(const float* __restrict__ in,
                                                     _Float16* __restrict__ out, int n) {
  int i = blockIdx.x * 256 + threadIdx.x;
  if (i < n) out[i] = (_Float16)in[i];
}

// in: [K][N] f32 row-major  ->  out: [N][K] f16 row-major
__global__ __launch_bounds__(256) void k_transpose_f16(const float* __restrict__ in,
                                                       _Float16* __restrict__ out,
                                                       int K, int N) {
  int i = blockIdx.x * 256 + threadIdx.x;
  if (i < K * N) {
    int k = i / N, n = i - k * N;
    out[n * K + k] = (_Float16)in[i];
  }
}

// ---------------------------------------------------------------------------
// Ping-pong double-buffered 32(M) x 64(N) x 1024(K) per-wave WMMA tile.
// Manual unroll-by-2: no register rotation copies.
static __device__ __forceinline__ void gemm_load(
    const _Float16* __restrict__ A, const _Float16* __restrict__ B, int k,
    v16h& a0, v16h& a1, v16h (&b)[4]) {
  a0 = load_frag_A_g(A + k, 1024);
  a1 = load_frag_A_g(A + 16 * 1024 + k, 1024);
#pragma unroll
  for (int nc = 0; nc < 4; ++nc) b[nc] = load_frag_B_g(B + nc * 16 * 1024 + k, 1024);
}

static __device__ __forceinline__ void gemm_mma(
    const v16h& a0, const v16h& a1, const v16h (&b)[4], v8f (&c)[2][4]) {
#pragma unroll
  for (int nc = 0; nc < 4; ++nc) {
    c[0][nc] = WMMA_F16(a0, b[nc], c[0][nc]);
    c[1][nc] = WMMA_F16(a1, b[nc], c[1][nc]);
  }
}

static __device__ __forceinline__ void gemm_tile_32x64(
    const _Float16* __restrict__ Abase,   // row-major, ld 1024, at row m0
    const _Float16* __restrict__ Bbase,   // row-major (N x K), ld 1024, at row n0
    v8f (&c)[2][4]) {
  v16h a0A, a1A, bA[4];
  v16h a0B, a1B, bB[4];
  gemm_load(Abase, Bbase, 0, a0A, a1A, bA);
  for (int k0 = 0; k0 < 1024; k0 += 64) {
    gemm_load(Abase, Bbase, k0 + 32, a0B, a1B, bB);   // k0+32 <= 992: always valid
    gemm_mma(a0A, a1A, bA, c);
    const int kn = (k0 + 64 < 1024) ? k0 + 64 : 0;    // last load is dead; clamp
    gemm_load(Abase, Bbase, kn, a0A, a1A, bA);
    gemm_mma(a0B, a1B, bB, c);
  }
}

// ---------------------------------------------------------------------------
// QKV GEMM: [4096 x 1024] @ w_attn^T-stored + bias, scattered into per-head
// Q [BH][S][64], K [BH][S][64], Vt [BH][64][S].
// Block = 8 waves = 64(M) x 256(N) tile; wave = 32 x 64.
__global__ __launch_bounds__(256)
void k_qkv_gemm(const _Float16* __restrict__ xh, const _Float16* __restrict__ wt,
                const float* __restrict__ bias,
                _Float16* __restrict__ qh, _Float16* __restrict__ kh,
                _Float16* __restrict__ vth) {
  const int L  = threadIdx.x & 31;
  const int wv = threadIdx.x >> 5;
  const int m0 = blockIdx.y * 64 + (wv >> 2) * 32;
  const int n0 = blockIdx.x * 256 + (wv & 3) * 64;

  v8f c[2][4];
#pragma unroll
  for (int g = 0; g < 2; ++g)
#pragma unroll
    for (int i = 0; i < 4; ++i) c[g][i] = (v8f){0,0,0,0,0,0,0,0};

  gemm_tile_32x64(xh + m0 * 1024, wt + n0 * 1024, c);

#pragma unroll
  for (int g = 0; g < 2; ++g) {
    const int rbase = m0 + g * 16 + ((L >> 4) << 3);
#pragma unroll
    for (int nc = 0; nc < 4; ++nc) {
      const int col = n0 + nc * 16 + (L & 15);
      const float bv = bias[col];
      const int which = col >> 10;      // 0=q, 1=k, 2=v
      const int cc = col & 1023;
      const int hh = cc >> 6;
      const int dd = cc & 63;
#pragma unroll
      for (int r = 0; r < 8; ++r) {
        const int row = rbase + r;
        const int bb = row >> 11, ss = row & 2047;
        const int hl = bb * 16 + hh;
        _Float16 hv = (_Float16)(c[g][nc][r] + bv);
        if (which == 0)      qh[(hl * 2048 + ss) * 64 + dd] = hv;
        else if (which == 1) kh[(hl * 2048 + ss) * 64 + dd] = hv;
        else                 vth[(hl * 64 + dd) * 2048 + ss] = hv;
      }
    }
  }
}

// ---------------------------------------------------------------------------
// Flash attention: one wave per 16-row Q tile, streaming 32-key blocks to the
// causal diagonal. V fragments issued early; next K fragments prefetched into
// alternating ping-pong buffers (no rotation copies). LDS only for the P-tile
// C->A layout shuffle, synced per-wave with s_wait_dscnt.
__global__ __launch_bounds__(256)
void k_attn(const _Float16* __restrict__ qh, const _Float16* __restrict__ kh,
            const _Float16* __restrict__ vth, _Float16* __restrict__ aout) {
  __shared__ __align__(16) _Float16 plds[8 * 512];   // 16x32 f16 per wave

  const int L  = threadIdx.x & 31;
  const int wv = threadIdx.x >> 5;
  const int w  = blockIdx.x * 8 + wv;
  const int head = w >> 7;            // / (S/16 = 128)
  const int qt   = w & 127;
  const int bb   = head >> 4;
  const int hh   = head & 15;
  const int qb   = qt << 4;

  const _Float16* Q  = qh  + head * (2048 * 64);
  const _Float16* K  = kh  + head * (2048 * 64);
  const _Float16* VT = vth + head * (64 * 2048);
  _Float16* pl = plds + wv * 512;

  const v16h aq0 = load_frag_A_g(Q + qb * 64, 64);       // d 0..31
  const v16h aq1 = load_frag_A_g(Q + qb * 64 + 32, 64);  // d 32..63

  v8f o[4];
#pragma unroll
  for (int i = 0; i < 4; ++i) o[i] = (v8f){0,0,0,0,0,0,0,0};
  float m[8], l[8];
#pragma unroll
  for (int r = 0; r < 8; ++r) { m[r] = -__builtin_inff(); l[r] = 0.0f; }

  const v8f zf = (v8f){0,0,0,0,0,0,0,0};
  const float scale = 0.125f;                  // 1/sqrt(64)
  const int rowb = qb + ((L >> 4) << 3);
  const int nb = (qb >> 5) + 1;                // causal block count

  // One key block: consume K fragments from bkc, prefetch block j+1 into bkp.
  auto block_step = [&](int j, v16h (&bkc)[4], v16h (&bkp)[4]) {
    const int kb = j << 5;

    // V fragments issued early: independent of scores/softmax.
    v16h bv[4];
#pragma unroll
    for (int nc = 0; nc < 4; ++nc)
      bv[nc] = load_frag_B_g(VT + nc * 16 * 2048 + kb, 2048);

    // Scores for 16 q-rows x 32 keys (waits only on bkc; bv stays in flight).
    v8f s0 = WMMA_F16(aq0, bkc[0], zf); s0 = WMMA_F16(aq1, bkc[1], s0);
    v8f s1 = WMMA_F16(aq0, bkc[2], zf); s1 = WMMA_F16(aq1, bkc[3], s1);

    // Prefetch next block's K fragments; they land during the softmax math.
    const int kbn = (j + 1 < nb) ? (j + 1) << 5 : kb;  // clamped (dead on last)
    bkp[0] = load_frag_B_g(K + kbn * 64, 64);
    bkp[1] = load_frag_B_g(K + kbn * 64 + 32, 64);
    bkp[2] = load_frag_B_g(K + (kbn + 16) * 64, 64);
    bkp[3] = load_frag_B_g(K + (kbn + 16) * 64 + 32, 64);

    const int col0 = kb + (L & 15);
    const int col1 = col0 + 16;
    float p0[8], p1[8], alpha[8];
#pragma unroll
    for (int r = 0; r < 8; ++r) {
      const int row = rowb + r;
      float a0 = (col0 <= row) ? s0[r] * scale : -1e9f;
      float a1 = (col1 <= row) ? s1[r] * scale : -1e9f;
      float bm = fmaxf(a0, a1);
      bm = fmaxf(bm, __shfl_xor(bm, 1));
      bm = fmaxf(bm, __shfl_xor(bm, 2));
      bm = fmaxf(bm, __shfl_xor(bm, 4));
      bm = fmaxf(bm, __shfl_xor(bm, 8));
      const float mn = fmaxf(m[r], bm);
      alpha[r] = __expf(m[r] - mn);
      m[r] = mn;
      p0[r] = __expf(a0 - mn);
      p1[r] = __expf(a1 - mn);
      float rs = p0[r] + p1[r];
      rs += __shfl_xor(rs, 1);
      rs += __shfl_xor(rs, 2);
      rs += __shfl_xor(rs, 4);
      rs += __shfl_xor(rs, 8);
      l[r] = l[r] * alpha[r] + rs;
    }
#pragma unroll
    for (int nc = 0; nc < 4; ++nc)
#pragma unroll
      for (int r = 0; r < 8; ++r) o[nc][r] *= alpha[r];

    // P tile f16 -> LDS row-major [16][32] (C layout -> A layout shuffle).
#pragma unroll
    for (int r = 0; r < 8; ++r) {
      const int rl = ((L >> 4) << 3) + r;
      pl[rl * 32 + (L & 15)]      = (_Float16)p0[r];
      pl[rl * 32 + (L & 15) + 16] = (_Float16)p1[r];
    }
    asm volatile("s_wait_dscnt 0x0" ::: "memory");   // LDS is in-order per wave

    const int arl = (L & 15) * 32 + ((L >> 4) << 3);
    v8h plo = *(const v8h*)(pl + arl);
    v8h phi = *(const v8h*)(pl + arl + 16);
    v16h ap = __builtin_shufflevector(plo, phi, 0,1,2,3,4,5,6,7,8,9,10,11,12,13,14,15);

    // O += P @ V  (waits for bv; prefetched K stays in flight).
#pragma unroll
    for (int nc = 0; nc < 4; ++nc) o[nc] = WMMA_F16(ap, bv[nc], o[nc]);

    asm volatile("" ::: "memory");   // keep next block's LDS stores below loads
  };

  // Prologue: K fragments for block 0 into buffer A, then alternate A/B.
  v16h bkA[4], bkB[4];
  bkA[0] = load_frag_B_g(K, 64);
  bkA[1] = load_frag_B_g(K + 32, 64);
  bkA[2] = load_frag_B_g(K + 16 * 64, 64);
  bkA[3] = load_frag_B_g(K + 16 * 64 + 32, 64);

  int j = 0;
  for (;;) {
    block_step(j, bkA, bkB);
    if (++j == nb) break;
    block_step(j, bkB, bkA);
    if (++j == nb) break;
  }

  float inv[8];
#pragma unroll
  for (int r = 0; r < 8; ++r) inv[r] = 1.0f / l[r];

#pragma unroll
  for (int nc = 0; nc < 4; ++nc)
#pragma unroll
    for (int r = 0; r < 8; ++r) {
      const int row = bb * 2048 + qb + ((L >> 4) << 3) + r;
      const int col = hh * 64 + nc * 16 + (L & 15);
      aout[row * 1024 + col] = (_Float16)(o[nc][r] * inv[r]);
    }
}

// ---------------------------------------------------------------------------
// Output projection: [4096 x 1024] @ [1024 x 1024] + bias -> f32 d_out
__global__ __launch_bounds__(256)
void k_proj_gemm(const _Float16* __restrict__ ah, const _Float16* __restrict__ wpt,
                 const float* __restrict__ bias, float* __restrict__ out) {
  const int L  = threadIdx.x & 31;
  const int wv = threadIdx.x >> 5;
  const int m0 = blockIdx.y * 64 + (wv >> 2) * 32;
  const int n0 = blockIdx.x * 256 + (wv & 3) * 64;

  v8f c[2][4];
#pragma unroll
  for (int g = 0; g < 2; ++g)
#pragma unroll
    for (int i = 0; i < 4; ++i) c[g][i] = (v8f){0,0,0,0,0,0,0,0};

  gemm_tile_32x64(ah + m0 * 1024, wpt + n0 * 1024, c);

#pragma unroll
  for (int g = 0; g < 2; ++g) {
    const int rbase = m0 + g * 16 + ((L >> 4) << 3);
#pragma unroll
    for (int nc = 0; nc < 4; ++nc) {
      const int col = n0 + nc * 16 + (L & 15);
      const float bv = bias[col];
#pragma unroll
      for (int r = 0; r < 8; ++r)
        out[(rbase + r) * 1024 + col] = c[g][nc][r] + bv;
    }
  }
}

// ---------------------------------------------------------------------------
extern "C" void kernel_launch(void* const* d_in, const int* in_sizes, int n_in,
                              void* d_out, int out_size, void* d_ws, size_t ws_size,
                              hipStream_t stream) {
  (void)in_sizes; (void)n_in; (void)out_size; (void)ws_size;
  const float* x      = (const float*)d_in[0];   // [2,2048,1024]
  const float* w_attn = (const float*)d_in[1];   // [1024,3072]
  const float* b_attn = (const float*)d_in[2];   // [3072]
  const float* w_proj = (const float*)d_in[3];   // [1024,1024]
  const float* b_proj = (const float*)d_in[4];   // [1024]
  float* out = (float*)d_out;

  // workspace layout (f16 elements); total ~48 MB
  _Float16* ws   = (_Float16*)d_ws;
  _Float16* xh   = ws;                  // 4096*1024
  _Float16* wt   = xh   + 4194304;      // 3072*1024 (w_attn^T)
  _Float16* qh   = wt   + 3145728;      // [BH][S][64]
  _Float16* kh   = qh   + 4194304;      // [BH][S][64]
  _Float16* vth  = kh   + 4194304;      // [BH][64][S]
  _Float16* aout = vth  + 4194304;      // [4096][1024]
  _Float16* wpt  = aout + 4194304;      // 1024*1024 (w_proj^T)

  k_cvt_f32_f16<<<4194304 / 256, 256, 0, stream>>>(x, xh, 4194304);
  k_transpose_f16<<<(1024 * 3072) / 256, 256, 0, stream>>>(w_attn, wt, 1024, 3072);
  k_transpose_f16<<<(1024 * 1024) / 256, 256, 0, stream>>>(w_proj, wpt, 1024, 1024);

  k_qkv_gemm<<<dim3(12, 64), 256, 0, stream>>>(xh, wt, b_attn, qh, kh, vth);
  k_attn<<<512, 256, 0, stream>>>(qh, kh, vth, aout);
  k_proj_gemm<<<dim3(4, 64), 256, 0, stream>>>(aout, wpt, b_proj, out);
}
